// TTDense_9706626090067
// MI455X (gfx1250) — matrix operand order repaired and checked by reading, not compiled, not run on hardware
//
#include <hip/hip_runtime.h>

// ---------------------------------------------------------------------------
// TT dense layer, fused, CDNA5 (gfx1250), wave32, bf16 WMMA w/ f32 accumulate.
//   step1: s1[m3*16+p1][m2*32+q1] = sum_m1 x[m1,m2,m3] * C0[m1,p1,q1]
//   step2: s2[(m3,p1)][(p2,q2)]   = sum_{m2,q1} s1 * C1[q1,m2,p2,q2]
//   step3: out[p1,p2,p3]          = relu(sum_{m3,q2} s2 * C2[q2,m3,p3] + bias)
// One block per batch element; s1 lives in LDS as pre-swizzled A-fragments.
// B1 slabs staged with CDNA5 async global->LDS (ASYNCcnt) instructions.
// ---------------------------------------------------------------------------

typedef __attribute__((ext_vector_type(16))) __bf16 v16bf;
typedef __attribute__((ext_vector_type(8)))  float  v8f;

// ws layout (ushort elements): B0 frags | B1 frags | B2 frags
#define WS_B0 0
#define WS_B1 16384                    // 32 nt * 32 lanes * 16
#define WS_B2 (16384 + 262144)         // 16 kt * 32 nt * 32 * 16
#define WS_TOTAL (16384 + 262144 + 8192)

// native f32 -> bf16 (RNE) store; backend emits a single v_cvt
__device__ __forceinline__ void st_bf16(unsigned short* p, float f) {
  *(__bf16*)p = (__bf16)f;
}

// ---- pre-pack cores to bf16 in WMMA B-fragment order --------------------
// B-frag (16x16x32): lane L holds n = base+L%16,
// k = kt*32 + (e/8)*16 + (L/16)*8 + e%8, e = 0..15 per lane (32 bytes).
__global__ void tt_prepack(const float* __restrict__ c0,
                           const float* __restrict__ c1,
                           const float* __restrict__ c2,
                           unsigned short* __restrict__ ws) {
  int t = blockIdx.x * blockDim.x + threadIdx.x;
  if (t < 16384) {
    // C0: K = m1 (16, zero-padded to 32), N = p1*32+q1 (512)
    int e = t & 15, L = (t >> 4) & 31, nt = t >> 9;
    int k = (e >> 3) * 16 + (L >> 4) * 8 + (e & 7);  // m1 or pad
    int n = nt * 16 + (L & 15);
    int p1 = n >> 5, q1 = n & 31;
    float v = (k < 16) ? c0[k * 512 + p1 * 32 + q1] : 0.0f;
    st_bf16(&ws[WS_B0 + t], v);
  } else if (t < 16384 + 262144) {
    // C1: K = m2*32+q1 (512 -> kt=m2), N = p2*32+q2 (512)
    int s = t - 16384;
    int e = s & 15, L = (s >> 4) & 31;
    int fi = s >> 9;                           // kt*32 + nt
    int nt = fi & 31, kt = fi >> 5;            // kt = m2
    int q1 = (e >> 3) * 16 + (L >> 4) * 8 + (e & 7);
    int n = nt * 16 + (L & 15);
    int p2 = n >> 5, q2 = n & 31;
    st_bf16(&ws[WS_B1 + s], c1[q1 * 8192 + kt * 512 + p2 * 32 + q2]);
  } else if (t < WS_TOTAL) {
    // C2: K = m3*32+q2 (512 -> kt3 = m3), N = p3 (16)
    int s = t - (16384 + 262144);
    int e = s & 15, L = (s >> 4) & 31, kt3 = s >> 9;
    int q2 = (e >> 3) * 16 + (L >> 4) * 8 + (e & 7);
    int p3 = L & 15;
    st_bf16(&ws[WS_B2 + s], c2[q2 * 256 + kt3 * 16 + p3]);
  }
}

// ---- main fused kernel ---------------------------------------------------
__global__ __launch_bounds__(256)
void tt_main(const float* __restrict__ x,
             const float* __restrict__ bias,
             const unsigned short* __restrict__ ws,
             float* __restrict__ out) {
  extern __shared__ char smem[];
  unsigned short* A1 = (unsigned short*)smem;                    // 256KB: s1 A-frags [rt2=m3][kt=m2] x 32 lanes x 16
  unsigned short* A2 = (unsigned short*)(smem + 262144);         // 16KB : s2 slab A-frags [kt3=m3] x 32 x 16
  unsigned short* Bb = (unsigned short*)(smem + 262144 + 16384); // 32KB : C1 slab frags
  float* outAcc = (float*)(smem + 262144 + 16384 + 32768);       // 1KB  : 16x16 f32

  const unsigned short* B0 = ws + WS_B0;
  const unsigned short* B1 = ws + WS_B1;
  const unsigned short* B2 = ws + WS_B2;

  const int b    = blockIdx.x;
  const int tid  = threadIdx.x;
  const int lane = tid & 31;
  const int w    = tid >> 5;        // wave 0..7
  const int cc   = lane & 15;
  const int hl   = lane >> 4;
  const int halfk = cc >> 3;        // scatter helpers (D-layout -> A-frag)
  const int ebase = cc & 7;

  // step-3 B fragments (C2) stay in registers for the whole kernel
  v16bf c2f0 = *(const v16bf*)(B2 + ((2 * w    ) * 32 + lane) * 16);
  v16bf c2f1 = *(const v16bf*)(B2 + ((2 * w + 1) * 32 + lane) * 16);

  // ---------------- Step 1: s1 = x * C0 (K=16 padded to 32) --------------
  union U16 { v16bf v; __bf16 h[16]; unsigned short u[16]; };
  U16 ax[2];
#pragma unroll
  for (int i = 0; i < 2; ++i) {
    int m2 = 2 * w + i;
#pragma unroll
    for (int e = 0; e < 8; ++e) {                 // A-frag: e<8 -> k = hl*8+e
      int m1 = hl * 8 + e;
      ax[i].h[e] = (__bf16)x[b * 4096 + m1 * 256 + m2 * 16 + cc];
    }
#pragma unroll
    for (int e = 8; e < 16; ++e) ax[i].u[e] = 0;  // k>=16 pad
  }

  for (int nt1 = 0; nt1 < 32; ++nt1) {
    v16bf bf = *(const v16bf*)(B0 + (nt1 * 32 + lane) * 16);
    int p1 = nt1 >> 1;
    int e  = (nt1 & 1) * 8 + ebase;               // q1 = (nt1&1)*16 + cc
#pragma unroll
    for (int i = 0; i < 2; ++i) {
      int m2 = 2 * w + i;
      v8f d = {0.f,0.f,0.f,0.f,0.f,0.f,0.f,0.f};
      d = __builtin_amdgcn_wmma_f32_16x16x32_bf16(false, ax[i].v, false, bf,
                                                  (short)0, d, false, false);
#pragma unroll
      for (int vv = 0; vv < 8; ++vv) {            // D row = m3 = vv + 8*hl
        int m3 = vv + 8 * hl;
        st_bf16(&A1[((m3 * 16 + m2) * 32 + halfk * 16 + p1) * 16 + e], d[vv]);
      }
    }
  }
  __syncthreads();

  // ---------------- Steps 2+3: loop over 16 column slabs (one p2 each) ---
  for (int p2 = 0; p2 < 16; ++p2) {
    outAcc[tid] = 0.0f;
    // stage C1 slab (32 frags, 32KB) from L2 into LDS with async-to-LDS DMA;
    // wave w copies 4 frags, 32B per lane (two b128, offset advances both
    // the global and LDS addresses per the ISA async pseudocode).
#pragma unroll
    for (int i = 0; i < 4; ++i) {
      int fi = w * 4 + i;
      int kt = fi >> 1, j = fi & 1;
      int nt = p2 * 2 + j;
      const unsigned short* src = B1 + ((kt * 32 + nt) * 32 + lane) * 16;
      unsigned short* dst = Bb + fi * 512 + lane * 16;
      unsigned lds_off = (unsigned)(unsigned long long)dst;   // low 32 bits = LDS offset
      unsigned long long ga = (unsigned long long)src;
      asm volatile("global_load_async_to_lds_b128 %0, %1, off\n\t"
                   "global_load_async_to_lds_b128 %0, %1, off offset:16"
                   :: "v"(lds_off), "v"(ga) : "memory");
    }
    if (p2 < 15)   // warm L2/near cache for next slab
      __builtin_prefetch(B1 + ((w * 2 * 32 + (p2 * 2 + 2)) * 32 + lane) * 16, 0, 1);
    asm volatile("s_wait_asynccnt 0" ::: "memory");
    __syncthreads();

    // step 2: wave computes 2 row-tiles (m3 = 2w, 2w+1) x 2 col-tiles, K=512
    v8f a00 = {0.f,0.f,0.f,0.f,0.f,0.f,0.f,0.f};
    v8f a01 = a00, a10 = a00, a11 = a00;
    for (int kt = 0; kt < 16; ++kt) {
      v16bf fa0 = *(const v16bf*)(A1 + (((2 * w    ) * 16 + kt) * 32 + lane) * 16);
      v16bf fa1 = *(const v16bf*)(A1 + (((2 * w + 1) * 16 + kt) * 32 + lane) * 16);
      v16bf fb0 = *(const v16bf*)(Bb + (kt * 2 + 0) * 512 + lane * 16);
      v16bf fb1 = *(const v16bf*)(Bb + (kt * 2 + 1) * 512 + lane * 16);
      a00 = __builtin_amdgcn_wmma_f32_16x16x32_bf16(false, fa0, false, fb0, (short)0, a00, false, false);
      a01 = __builtin_amdgcn_wmma_f32_16x16x32_bf16(false, fa0, false, fb1, (short)0, a01, false, false);
      a10 = __builtin_amdgcn_wmma_f32_16x16x32_bf16(false, fa1, false, fb0, (short)0, a10, false, false);
      a11 = __builtin_amdgcn_wmma_f32_16x16x32_bf16(false, fa1, false, fb1, (short)0, a11, false, false);
    }
    // scatter s2 tiles (bf16) into step-3 A-fragment layout:
    // addr = (m3*32 + halfk*16 + p1)*16 + (j*8 + cc%8), q2 = j*16 + cc
    {
      int e0 = ebase, e1 = 8 + ebase;
      int m3a = 2 * w, m3b = 2 * w + 1;
#pragma unroll
      for (int vv = 0; vv < 8; ++vv) {
        int p1v = vv + 8 * hl;                    // D row = p1
        st_bf16(&A2[(m3a * 32 + halfk * 16 + p1v) * 16 + e0], a00[vv]);
        st_bf16(&A2[(m3a * 32 + halfk * 16 + p1v) * 16 + e1], a01[vv]);
        st_bf16(&A2[(m3b * 32 + halfk * 16 + p1v) * 16 + e0], a10[vv]);
        st_bf16(&A2[(m3b * 32 + halfk * 16 + p1v) * 16 + e1], a11[vv]);
      }
    }
    __syncthreads();

    // step 3: contract (m3,q2); wave w handles kt3 = 2w, 2w+1 then reduces
    {
      v16bf fa0 = *(const v16bf*)(A2 + ((2 * w    ) * 32 + lane) * 16);
      v16bf fa1 = *(const v16bf*)(A2 + ((2 * w + 1) * 32 + lane) * 16);
      v8f acc3 = {0.f,0.f,0.f,0.f,0.f,0.f,0.f,0.f};
      acc3 = __builtin_amdgcn_wmma_f32_16x16x32_bf16(false, fa0, false, c2f0, (short)0, acc3, false, false);
      acc3 = __builtin_amdgcn_wmma_f32_16x16x32_bf16(false, fa1, false, c2f1, (short)0, acc3, false, false);
#pragma unroll
      for (int vv = 0; vv < 8; ++vv) {
        int p1v = vv + 8 * hl;                    // D: row p1, col p3 = cc
        atomicAdd(&outAcc[p1v * 16 + cc], acc3[vv]);
      }
    }
    __syncthreads();

    // bias + relu + store (thread t -> p1 = t/16, p3 = t%16)
    {
      int oidx = (tid >> 4) * 256 + p2 * 16 + (tid & 15);
      float v = outAcc[tid] + bias[oidx];
      out[b * 4096 + oidx] = v > 0.0f ? v : 0.0f;
    }
    __syncthreads();
  }
}

extern "C" void kernel_launch(void* const* d_in, const int* in_sizes, int n_in,
                              void* d_out, int out_size, void* d_ws, size_t ws_size,
                              hipStream_t stream) {
  const float* x    = (const float*)d_in[0];
  const float* c0   = (const float*)d_in[1];
  const float* c1   = (const float*)d_in[2];
  const float* c2   = (const float*)d_in[3];
  const float* bias = (const float*)d_in[4];
  float* out = (float*)d_out;
  unsigned short* ws = (unsigned short*)d_ws;   // uses 573440 bytes

  tt_prepack<<<(WS_TOTAL + 255) / 256, 256, 0, stream>>>(c0, c1, c2, ws);

  const size_t shmem = 262144 + 16384 + 32768 + 1024;   // 305 KB < 320 KB/WGP
  (void)hipFuncSetAttribute((const void*)tt_main,
                            hipFuncAttributeMaxDynamicSharedMemorySize,
                            (int)shmem);
  tt_main<<<1024, 256, shmem, stream>>>(x, bias, ws, out);
}
